// FourierLayer_66340064854413
// MI455X (gfx1250) — matrix operand order repaired and checked by reading, not compiled, not run on hardware
//
#include <hip/hip_runtime.h>
#include <hip/hip_bf16.h>

#define D_MODEL 512
#define MODES   16
#define BATCH   8
#define SEQ     4096
#define EPS     1e-5f

#if __has_builtin(__builtin_amdgcn_global_load_async_to_lds_b128) && \
    __has_builtin(__builtin_amdgcn_s_wait_asynccnt)
#define USE_ASYNC_LDS 1
#else
#define USE_ASYNC_LDS 0
#endif

typedef __attribute__((ext_vector_type(16))) __bf16 v16bf;
typedef __attribute__((ext_vector_type(8)))  __bf16 v8bf;
typedef __attribute__((ext_vector_type(8)))  float  v8f;
typedef __attribute__((ext_vector_type(4)))  int    v4i;
typedef __attribute__((address_space(1))) v4i as1_v4i;
typedef __attribute__((address_space(3))) v4i as3_v4i;

// ---- WMMA operand builders (per CDNA5 ISA 7.12.2 layouts, wave32) ----
// A (16x32 bf16): lane l<16 : M=l,    elems 0..7 = K0..7,  8..15 = K16..23
//                 lane l>=16: M=l-16, elems 0..7 = K8..15, 8..15 = K24..31
static __device__ inline v16bf load_a(const __bf16* row_base, int lane) {
  const int koff = (lane >> 4) * 8;
  v8bf lo = *(const v8bf*)(row_base + koff);
  v8bf hi = *(const v8bf*)(row_base + koff + 16);
  v16bf r;
#pragma unroll
  for (int i = 0; i < 8; i++) { r[i] = lo[i]; r[i + 8] = hi[i]; }
  return r;
}
// B (32x16 bf16) loaded from B^T (row-major [N][K]):
// lane l<16 : N=l, K0..7,K16..23 ; lane l>=16 : N=l-16, K8..15,K24..31
static __device__ inline v16bf load_bt(const __bf16* bt_row_k0, int lane) {
  const __bf16* p = bt_row_k0 + (lane >> 4) * 8;
  v8bf lo = *(const v8bf*)p;
  v8bf hi = *(const v8bf*)(p + 16);
  v16bf r;
#pragma unroll
  for (int i = 0; i < 8; i++) { r[i] = lo[i]; r[i + 8] = hi[i]; }
  return r;
}
// Same B^T pattern but from LDS, as 2x16B loads (16B-aligned rows of 136 elems)
static __device__ inline v16bf load_bt_lds(const __bf16* bt_row_k0, int lane) {
  const __bf16* p = bt_row_k0 + (lane >> 4) * 8;
  v8bf lo = *(const v8bf*)p;
  v8bf hi = *(const v8bf*)(p + 16);
  v16bf r;
#pragma unroll
  for (int i = 0; i < 8; i++) { r[i] = lo[i]; r[i + 8] = hi[i]; }
  return r;
}

static __device__ inline float wave_red(float v) {
#pragma unroll
  for (int off = 16; off; off >>= 1) v += __shfl_xor(v, off, 32);
  return v;
}

// ---- prep: forward basis FB[32][SEQ] + combined inverse basis IBC[SEQ][128] ----
// IBC[s][0..31]     = IB[s][*]                  (direct xs term)
// IBC[s][32+t*32+j] = IB[s+t-1][j], 0 if OOB    (conv taps, low-rank form)
__global__ void k_prep_basis(__bf16* __restrict__ FB, __bf16* __restrict__ IBC) {
  const int tid = blockIdx.x * blockDim.x + threadIdx.x;  // 65536 = 16*4096
  const int k = tid >> 12;
  const int s = tid & (SEQ - 1);
  const int m = (k * s) & (SEQ - 1);
  const float th = 6.28318530717958647692f * (float)m / (float)SEQ;
  float sn, cs;
  __sincosf(th, &sn, &cs);
  FB[(2 * k) * SEQ + s]     = (__bf16)cs;      // F_r row
  FB[(2 * k + 1) * SEQ + s] = (__bf16)(-sn);   // F_i row
  const float scale = (k == 0) ? (1.0f / (float)SEQ) : (2.0f / (float)SEQ);
  const __bf16 cr = (__bf16)(scale * cs);
  const __bf16 ci = (__bf16)(-scale * sn);
  IBC[(size_t)s * 128 + 2 * k]     = cr;
  IBC[(size_t)s * 128 + 2 * k + 1] = ci;
#pragma unroll
  for (int t = 0; t < 3; t++) {
    const int sd = s + 1 - t;  // IBC[sd][32+t*32+j] = IB[sd+t-1][j] = IB[s][j]
    if (sd >= 0 && sd < SEQ) {
      IBC[(size_t)sd * 128 + 32 + t * 32 + 2 * k]     = cr;
      IBC[(size_t)sd * 128 + 32 + t * 32 + 2 * k + 1] = ci;
    }
  }
  if (s == 0) {  // row 0, tap t=0 references IB[-1] -> zero
    IBC[32 + 2 * k] = (__bf16)0.0f;
    IBC[33 + 2 * k] = (__bf16)0.0f;
  }
  if (s == SEQ - 1) {  // last row, tap t=2 references IB[SEQ] -> zero
    IBC[(size_t)(SEQ - 1) * 128 + 96 + 2 * k] = (__bf16)0.0f;
    IBC[(size_t)(SEQ - 1) * 128 + 97 + 2 * k] = (__bf16)0.0f;
  }
}

// ---- prep: conv weights (oc,ic,3) -> bf16 [tap][oc][ic] ----
__global__ void k_prep_w(const float* __restrict__ conv_w, __bf16* __restrict__ Wb) {
  const int tid = blockIdx.x * blockDim.x + threadIdx.x;  // 3*512*512
  const int t = tid / (D_MODEL * D_MODEL);
  const int r = tid % (D_MODEL * D_MODEL);
  const int oc = r / D_MODEL, ic = r % D_MODEL;
  Wb[tid] = (__bf16)conv_w[(oc * D_MODEL + ic) * 3 + t];  // target idx == tid
}

// ---- fused LayerNorm + transpose: x[b][s][d] -> xnT[b][d][s] (bf16) ----
__global__ void k_ln_t(const float* __restrict__ x, const float* __restrict__ gamma,
                       const float* __restrict__ beta, __bf16* __restrict__ xnT) {
  __shared__ __bf16 tileT[D_MODEL][34];  // [d][s_local], 17-dword odd stride
  const int lane = threadIdx.x & 31;
  const int wv = threadIdx.x >> 5;
  const int b = blockIdx.y;
  const int s0 = blockIdx.x * 32;
#pragma unroll
  for (int r = 0; r < 4; r++) {
    const int sl = wv * 4 + r;
    const float* xr = x + ((size_t)b * SEQ + s0 + sl) * D_MODEL;
    float v[16];
    float s1 = 0.f, s2 = 0.f;
#pragma unroll
    for (int j = 0; j < 16; j++) {
      v[j] = xr[j * 32 + lane];
      s1 += v[j];
      s2 += v[j] * v[j];
    }
    s1 = wave_red(s1);
    s2 = wave_red(s2);
    const float mu = s1 * (1.0f / D_MODEL);
    const float var = s2 * (1.0f / D_MODEL) - mu * mu;
    const float rs = __frsqrt_rn(var + EPS);
#pragma unroll
    for (int j = 0; j < 16; j++) {
      const int d = j * 32 + lane;
      tileT[d][sl] = (__bf16)((v[j] - mu) * rs * gamma[d] + beta[d]);
    }
  }
  __syncthreads();
  for (int d = wv; d < D_MODEL; d += 8)  // 64-byte coalesced rows
    xnT[((size_t)b * D_MODEL + d) * SEQ + s0 + lane] = tileT[d][lane];
}

// ---- forward spectral GEMM + fused complex mode-mix ----
// F = FB[32][S] @ xn[b][S][512]; then c = F * w written to G (row major) and
// CT[b][d][0..31].  Both m-tiles per wave: B operand reused for 2 WMMAs.
__global__ void k_fwd(const __bf16* __restrict__ FB, const __bf16* __restrict__ xnT,
                      const float* __restrict__ wr, const float* __restrict__ wi,
                      __bf16* __restrict__ G, __bf16* __restrict__ CT) {
  const int lane = threadIdx.x & 31;
  const int wv = threadIdx.x >> 5;
  const int d0 = (blockIdx.x * 8 + wv) * 16;
  const int b = blockIdx.y;
  const int row = lane & 15, col = lane & 15, half = lane >> 4;
  v8f acc0 = {}, acc1 = {};
  const __bf16* arow0 = FB + (size_t)row * SEQ;
  const __bf16* arow1 = FB + (size_t)(16 + row) * SEQ;
  const __bf16* btrow = xnT + ((size_t)b * D_MODEL + d0 + col) * SEQ;
  for (int k0 = 0; k0 < SEQ; k0 += 32) {
    v16bf bm = load_bt(btrow + k0, lane);
    v16bf a0 = load_a(arow0 + k0, lane);
    acc0 = __builtin_amdgcn_wmma_f32_16x16x32_bf16(false, a0, false, bm, (short)0,
                                                   acc0, false, false);
    v16bf a1 = load_a(arow1 + k0, lane);
    acc1 = __builtin_amdgcn_wmma_f32_16x16x32_bf16(false, a1, false, bm, (short)0,
                                                   acc1, false, false);
  }
  const int rbase = half * 8;
  const int d = d0 + col;
  __bf16* Gb = G + (size_t)b * 32 * D_MODEL;
  __bf16* CTr = CT + ((size_t)b * D_MODEL + d) * 128;
#pragma unroll
  for (int j = 0; j < 4; j++) {
    // acc0: rows m = rbase+2j (F_r), rbase+2j+1 (F_i); mode k = rbase/2 + j
    {
      const int k = half * 4 + j;
      const float Fr = acc0[2 * j], Fi = acc0[2 * j + 1];
      const float wrv = wr[d * MODES + k], wiv = wi[d * MODES + k];
      const float cr = Fr * wrv - Fi * wiv;
      const float ci = Fr * wiv + Fi * wrv;
      const int mm = rbase + 2 * j;
      Gb[(size_t)mm * D_MODEL + d]       = (__bf16)cr;
      Gb[(size_t)(mm + 1) * D_MODEL + d] = (__bf16)ci;
      CTr[mm]     = (__bf16)cr;
      CTr[mm + 1] = (__bf16)ci;
    }
    // acc1: rows m = 16+rbase+2j; mode k = 8 + rbase/2 + j
    {
      const int k = 8 + half * 4 + j;
      const float Fr = acc1[2 * j], Fi = acc1[2 * j + 1];
      const float wrv = wr[d * MODES + k], wiv = wi[d * MODES + k];
      const float cr = Fr * wrv - Fi * wiv;
      const float ci = Fr * wiv + Fi * wrv;
      const int mm = 16 + rbase + 2 * j;
      Gb[(size_t)mm * D_MODEL + d]       = (__bf16)cr;
      Gb[(size_t)(mm + 1) * D_MODEL + d] = (__bf16)ci;
      CTr[mm]     = (__bf16)cr;
      CTr[mm + 1] = (__bf16)ci;
    }
  }
}

// ---- tap projection: H_t = G @ W_t^T -> CT[b][oc][32+t*32+m] ----
__global__ void k_h(const __bf16* __restrict__ G, const __bf16* __restrict__ Wb,
                    __bf16* __restrict__ CT) {
  const int lane = threadIdx.x & 31;
  const int wv = threadIdx.x >> 5;
  const int oc0 = (blockIdx.x * 8 + wv) * 16;
  const int m0 = blockIdx.y * 16;
  const int b = blockIdx.z / 3;
  const int t = blockIdx.z % 3;
  const int row = lane & 15, col = lane & 15;
  v8f acc = {};
  const __bf16* arow = G + ((size_t)b * 32 + m0 + row) * D_MODEL;
  const __bf16* btrow = Wb + ((size_t)t * D_MODEL + oc0 + col) * D_MODEL;
  for (int k0 = 0; k0 < D_MODEL; k0 += 32) {
    v16bf a = load_a(arow + k0, lane);
    v16bf bm = load_bt(btrow + k0, lane);
    acc = __builtin_amdgcn_wmma_f32_16x16x32_bf16(false, a, false, bm, (short)0, acc,
                                                  false, false);
  }
  const int rbase = (lane >> 4) * 8;
#pragma unroll
  for (int v = 0; v < 8; v++)
    CT[((size_t)b * D_MODEL + oc0 + col) * 128 + 32 + t * 32 + m0 + rbase + v] =
        (__bf16)acc[v];
}

// ---- final fused GEMM: out = x + bias + IBC[S x 128] @ CK[b][128 x 512] ----
// Block stages shared CT slice (64 d-rows x 128) in LDS via async-to-LDS;
// each wave: 16 s-rows x 64 d-cols (A reused for 4 WMMAs per k-step).
__global__ void k_final(const __bf16* __restrict__ IBC, const __bf16* __restrict__ CT,
                        const float* __restrict__ x, const float* __restrict__ bias,
                        float* __restrict__ out) {
  __shared__ __bf16 tileB[64][136];  // row stride 68 dwords (conflict-friendly)
  const int lane = threadIdx.x & 31;
  const int wv = threadIdx.x >> 5;
  const int s0 = (blockIdx.y * 8 + wv) * 16;
  const int d0 = blockIdx.x * 64;
  const int b = blockIdx.z;
  const int row = lane & 15, col = lane & 15;

  // cooperative stage of CT[b][d0..d0+63][0..127] -> LDS
  const __bf16* CTs = CT + ((size_t)b * D_MODEL + d0) * 128;
#pragma unroll
  for (int i = 0; i < 4; i++) {
    const int ci = threadIdx.x * 4 + i;      // 1024 chunks of 8 elems (16B)
    const int r = ci >> 4, ch = ci & 15;
#if USE_ASYNC_LDS
    __builtin_amdgcn_global_load_async_to_lds_b128(
        (as1_v4i*)(CTs + r * 128 + ch * 8),
        (as3_v4i*)(&tileB[r][ch * 8]), 0, 0);
#else
    *(v8bf*)(&tileB[r][ch * 8]) = *(const v8bf*)(CTs + r * 128 + ch * 8);
#endif
  }
#if USE_ASYNC_LDS
  __builtin_amdgcn_s_wait_asynccnt(0);
#endif
  __syncthreads();

  v8f acc[4] = {{}, {}, {}, {}};
  const __bf16* arow = IBC + (size_t)(s0 + row) * 128;
#pragma unroll
  for (int k0 = 0; k0 < 128; k0 += 32) {
    v16bf a = load_a(arow + k0, lane);
#pragma unroll
    for (int u = 0; u < 4; u++) {
      v16bf bm = load_bt_lds(&tileB[u * 16 + col][k0], lane);
      acc[u] = __builtin_amdgcn_wmma_f32_16x16x32_bf16(false, a, false, bm, (short)0,
                                                       acc[u], false, false);
    }
  }
  const int rbase = (lane >> 4) * 8;
#pragma unroll
  for (int u = 0; u < 4; u++) {
    const int d = d0 + u * 16 + col;
    const float bs = bias[d];
#pragma unroll
    for (int v = 0; v < 8; v++) {
      const size_t idx = ((size_t)b * SEQ + s0 + rbase + v) * D_MODEL + d;
      __builtin_prefetch(&x[idx + 8 * D_MODEL], 0, 1);
      out[idx] = acc[u][v] + bs + x[idx];
    }
  }
}

extern "C" void kernel_launch(void* const* d_in, const int* in_sizes, int n_in,
                              void* d_out, int out_size, void* d_ws, size_t ws_size,
                              hipStream_t stream) {
  const float* x      = (const float*)d_in[0];
  const float* gamma  = (const float*)d_in[1];
  const float* beta   = (const float*)d_in[2];
  const float* wr     = (const float*)d_in[3];
  const float* wi     = (const float*)d_in[4];
  const float* conv_w = (const float*)d_in[5];
  const float* conv_b = (const float*)d_in[6];
  float* out = (float*)d_out;

  char* ws = (char*)d_ws;
  size_t off = 0;
  __bf16* FB  = (__bf16*)(ws + off); off += (size_t)32 * SEQ * 2;               // 256 KB
  __bf16* IBC = (__bf16*)(ws + off); off += (size_t)SEQ * 128 * 2;              // 1 MB
  __bf16* Wb  = (__bf16*)(ws + off); off += (size_t)3 * D_MODEL * D_MODEL * 2;  // 1.5 MB
  __bf16* xnT = (__bf16*)(ws + off); off += (size_t)BATCH * D_MODEL * SEQ * 2;  // 32 MB
  __bf16* G   = (__bf16*)(ws + off); off += (size_t)BATCH * 32 * D_MODEL * 2;   // 256 KB
  __bf16* CT  = (__bf16*)(ws + off); off += (size_t)BATCH * D_MODEL * 128 * 2;  // 1 MB

  hipLaunchKernelGGL(k_prep_basis, dim3(256), dim3(256), 0, stream, FB, IBC);
  hipLaunchKernelGGL(k_prep_w, dim3(3072), dim3(256), 0, stream, conv_w, Wb);
  hipLaunchKernelGGL(k_ln_t, dim3(SEQ / 32, BATCH), dim3(256), 0, stream,
                     x, gamma, beta, xnT);
  hipLaunchKernelGGL(k_fwd, dim3(D_MODEL / 128, BATCH), dim3(256), 0, stream,
                     FB, xnT, wr, wi, G, CT);
  hipLaunchKernelGGL(k_h, dim3(D_MODEL / 128, 2, 3 * BATCH), dim3(256), 0, stream,
                     G, Wb, CT);
  hipLaunchKernelGGL(k_final, dim3(D_MODEL / 64, SEQ / 128, BATCH), dim3(256), 0,
                     stream, IBC, CT, x, conv_b, out);
}
